// ConsciousLM_30185030157066
// MI455X (gfx1250) — matrix operand order repaired and checked by reading, not compiled, not run on hardware
//
#include <hip/hip_runtime.h>
#include <cstdint>

// ---------------- model constants ----------------
#define B_   4
#define T_   1024
#define D_   768
#define H_   12
#define L_   6
#define R_   8
#define V_   256
#define HD_  64
#define M_   (B_ * T_)          // 4096 rows (B*T)
#define D3_  (3 * D_)           // 2304
#define D4_  (4 * D_)           // 3072
#define GATE_ 0.001f
#define EPS_  1e-5f

typedef __attribute__((ext_vector_type(16))) __bf16        v16bf;
typedef __attribute__((ext_vector_type(2)))  __bf16        v2bf;
typedef __attribute__((ext_vector_type(8)))  float         v8f;
typedef __attribute__((ext_vector_type(8)))  unsigned int  v8u;

// two f32 -> packed bf16 dword (compiler should emit v_cvt_pk_bf16_f32)
__device__ __forceinline__ unsigned pk2(float x, float y) {
  v2bf p;
  p[0] = (__bf16)x;
  p[1] = (__bf16)y;
  return __builtin_bit_cast(unsigned, p);
}

// ---------------------------------------------------------------------------
// Generic batched GEMM on the WMMA pipe.
//   TRANSB==0 :  C[m,n] = A[m,:] . W[n,:]   (W row-major N x K, i.e. A @ W^T)
//   TRANSB==1 :  C[m,n] = A[m,:] . W[:,n]   (W row-major K x N, i.e. A @ W)
//   epilogue :  v = gemm + bias[n]; if ACT v=gelu(v);
//               C = (accum ? C : 0) + alpha * rowscale[m] * v
//
// Block: 256 thr = 8 wave32, waves 2(M) x 4(N); each wave owns a 32x32 output
// (2x2 register tile = 4 WMMA per K-step) -> block tile 64(M) x 128(N),
// K-step 32 (one v_wmma_f32_16x16x32_bf16 per 16x16 sub-tile).
// A tile (64x32) staged f32->bf16 through LDS with b128 ds traffic;
// B fragments built per lane from global with b128 loads (TRANSB=0).
// Requires M%64==0, K%32==0; N may be any multiple of 16 (cols clamped/guarded).
// ---------------------------------------------------------------------------
template <int ACT, int TRANSB>
__global__ void __launch_bounds__(256)
k_gemm(const float* __restrict__ A, long lda, long sA,
       const float* __restrict__ W, long ldw, long sW,
       const float* __restrict__ bias,
       float* __restrict__ C, long ldc, long sC,
       int M, int N, int K, int accum, float alpha,
       const float* __restrict__ rowscale, int rsStride)
{
  __shared__ unsigned ldsA[64 * 20];   // 64 rows x 16 bf16-pair dwords + pad 4

  const int tid  = threadIdx.x;
  const int lane = tid & 31;
  const int wid  = tid >> 5;
  const int wm   = wid >> 2;           // wave row 0..1  (32 rows each)
  const int wn   = wid & 3;            // wave col 0..3  (32 cols each)
  const int m0   = blockIdx.y * 64;
  const int n0   = blockIdx.x * 128;
  const int z    = blockIdx.z;

  A += (long)z * sA;  W += (long)z * sW;  C += (long)z * sC;

  const int l15  = lane & 15;
  const int half = lane >> 4;

  // this wave's two output column groups (clamped for address safety when N<tile)
  int ngl[2];
#pragma unroll
  for (int ni = 0; ni < 2; ++ni) {
    int c = n0 + wn * 32 + ni * 16 + l15;
    ngl[ni] = (c < N) ? c : (N - 1);
  }

  v8f acc[2][2];
#pragma unroll
  for (int mi = 0; mi < 2; ++mi)
#pragma unroll
    for (int ni = 0; ni < 2; ++ni)
#pragma unroll
      for (int i = 0; i < 8; ++i) acc[mi][ni][i] = 0.0f;

  const int t8   = tid * 8;            // each thread stages 8 consecutive floats
  const int arow = t8 >> 5;            // 0..63
  const int acol = t8 & 31;            // 0,8,16,24

  for (int kb = 0; kb < K; kb += 32) {
    // ---- cooperative A stage: 64x32 f32 -> packed bf16 pairs in LDS ----
    {
      const float* ap = &A[(long)(m0 + arow) * lda + kb + acol];
      const float4 a0 = *reinterpret_cast<const float4*>(ap);
      const float4 a1 = *reinterpret_cast<const float4*>(ap + 4);
      uint4 pk;
      pk.x = pk2(a0.x, a0.y);  pk.y = pk2(a0.z, a0.w);
      pk.z = pk2(a1.x, a1.y);  pk.w = pk2(a1.z, a1.w);
      *reinterpret_cast<uint4*>(&ldsA[arow * 20 + (acol >> 1)]) = pk;
    }
    __syncthreads();

    // ---- A fragments (16x32 bf16 each; per-lane dwords = two runs of 4) ----
    v16bf afr[2];
#pragma unroll
    for (int mi = 0; mi < 2; ++mi) {
      const unsigned* ar = &ldsA[(wm * 32 + mi * 16 + l15) * 20 + half * 4];
      const uint4 u0 = *reinterpret_cast<const uint4*>(ar);
      const uint4 u1 = *reinterpret_cast<const uint4*>(ar + 8);
      v8u au;
      au[0] = u0.x; au[1] = u0.y; au[2] = u0.z; au[3] = u0.w;
      au[4] = u1.x; au[5] = u1.y; au[6] = u1.z; au[7] = u1.w;
      afr[mi] = __builtin_bit_cast(v16bf, au);
    }

    // ---- B fragments (32x16 bf16: lanes 0-15 K=0..7,16..23, lanes 16-31 K=8..15,24..31) --
    v16bf bfr[2];
#pragma unroll
    for (int ni = 0; ni < 2; ++ni) {
      float f[16];
      if (TRANSB == 0) {
        const float* wp = &W[(long)ngl[ni] * ldw + kb + half * 8];
        const float4 f0 = *reinterpret_cast<const float4*>(wp);
        const float4 f1 = *reinterpret_cast<const float4*>(wp + 4);
        const float4 f2 = *reinterpret_cast<const float4*>(wp + 16);
        const float4 f3 = *reinterpret_cast<const float4*>(wp + 20);
        f[0]=f0.x; f[1]=f0.y; f[2]=f0.z; f[3]=f0.w;
        f[4]=f1.x; f[5]=f1.y; f[6]=f1.z; f[7]=f1.w;
        f[8]=f2.x; f[9]=f2.y; f[10]=f2.z; f[11]=f2.w;
        f[12]=f3.x; f[13]=f3.y; f[14]=f3.z; f[15]=f3.w;
      } else {
#pragma unroll
        for (int j = 0; j < 16; ++j)
          f[j] = W[(long)(kb + half * 8 + (j & 7) + (j >> 3) * 16) * ldw + ngl[ni]];
      }
      v16bf bv;
#pragma unroll
      for (int j = 0; j < 16; ++j) bv[j] = (__bf16)f[j];
      bfr[ni] = bv;
    }
    if (TRANSB == 0 && kb + 32 < K)    // pull next K-slice of B toward caches
      __builtin_prefetch(&W[(long)ngl[0] * ldw + kb + 32], 0, 0);

    // ---- 4 WMMAs per wave per K-step ----
#pragma unroll
    for (int mi = 0; mi < 2; ++mi)
#pragma unroll
      for (int ni = 0; ni < 2; ++ni)
        acc[mi][ni] = __builtin_amdgcn_wmma_f32_16x16x32_bf16(
            false, afr[mi], false, bfr[ni], (short)0, acc[mi][ni], false, false);
    __syncthreads();
  }

  // ---- epilogue: C/D layout VGPR r -> M = r + 8*half, N = lane&15 ----
#pragma unroll
  for (int mi = 0; mi < 2; ++mi) {
#pragma unroll
    for (int ni = 0; ni < 2; ++ni) {
      const int col = n0 + wn * 32 + ni * 16 + l15;
      if (col < N) {
#pragma unroll
        for (int r = 0; r < 8; ++r) {
          const int row = m0 + wm * 32 + mi * 16 + r + half * 8;
          float v = acc[mi][ni][r];
          if (bias) v += bias[col];
          if (ACT == 1) v = 0.5f * v * (1.0f + erff(v * 0.70710678118f));
          float sc = alpha;
          if (rowscale) sc *= rowscale[(long)row * rsStride];
          float* p = &C[(long)row * ldc + col];
          *p = (accum ? *p : 0.0f) + sc * v;
        }
      }
    }
  }
}

// ---------------- elementwise / reduction kernels ----------------

__global__ void k_embed(const int* __restrict__ idx, const float* __restrict__ tok,
                        const float* __restrict__ pos, float* __restrict__ x)
{
  long i = (long)blockIdx.x * blockDim.x + threadIdx.x;
  if (i >= (long)M_ * D_) return;
  long row = i / D_; int d = (int)(i % D_); int t = (int)(row % T_);
  x[i] = tok[(long)idx[row] * D_ + d] + pos[(long)t * D_ + d];
}

__global__ void __launch_bounds__(256)
k_layernorm(const float* __restrict__ in, float* __restrict__ out,
            const float* __restrict__ w, const float* __restrict__ b, int Dd)
{
  __shared__ float red[256];
  const int row = blockIdx.x, tid = threadIdx.x;
  const float* x = in + (long)row * Dd;
  float s = 0.f, ss = 0.f;
  for (int d = tid; d < Dd; d += 256) { float v = x[d]; s += v; ss += v * v; }
  red[tid] = s; __syncthreads();
  for (int o = 128; o > 0; o >>= 1) { if (tid < o) red[tid] += red[tid + o]; __syncthreads(); }
  const float mu = red[0] / Dd; __syncthreads();
  red[tid] = ss; __syncthreads();
  for (int o = 128; o > 0; o >>= 1) { if (tid < o) red[tid] += red[tid + o]; __syncthreads(); }
  const float var = red[0] / Dd - mu * mu;
  const float rstd = rsqrtf(var + EPS_);
  float* y = out + (long)row * Dd;
  for (int d = tid; d < Dd; d += 256) y[d] = (x[d] - mu) * rstd * w[d] + b[d];
}

__global__ void __launch_bounds__(256) k_softmax_causal(float* __restrict__ s)
{
  __shared__ float red[256];
  const int q = blockIdx.x, h = blockIdx.y, tid = threadIdx.x;
  float* row = s + ((long)h * T_ + q) * (long)T_;
  const float scale = 0.125f;  // 1/sqrt(64)
  float m = -3.4e38f;
  for (int k = tid; k <= q; k += 256) m = fmaxf(m, row[k] * scale);
  red[tid] = m; __syncthreads();
  for (int o = 128; o > 0; o >>= 1) { if (tid < o) red[tid] = fmaxf(red[tid], red[tid + o]); __syncthreads(); }
  m = red[0]; __syncthreads();
  float sum = 0.f;
  for (int k = tid; k <= q; k += 256) sum += expf(row[k] * scale - m);
  red[tid] = sum; __syncthreads();
  for (int o = 128; o > 0; o >>= 1) { if (tid < o) red[tid] += red[tid + o]; __syncthreads(); }
  const float inv = 1.0f / red[0];
  for (int k = tid; k < T_; k += 256)
    row[k] = (k <= q) ? expf(row[k] * scale - m) * inv : 0.0f;
}

__global__ void k_build_nb(const float* __restrict__ x, float* __restrict__ nb)
{
  long i = (long)blockIdx.x * blockDim.x + threadIdx.x;
  if (i >= (long)M_ * D3_) return;
  long row = i / D3_; int c = (int)(i % D3_);
  int part = c / D_, d = c % D_;
  int bb = (int)(row / T_), t = (int)(row % T_);
  int ts = t + part - 1; ts = ts < 0 ? 0 : (ts >= T_ ? T_ - 1 : ts);
  nb[i] = x[((long)bb * T_ + ts) * D_ + d];
}

// softmax(x @ W_rw^T + b_rw) over R=8; 8 adjacent lanes per row
__global__ void __launch_bounds__(256)
k_probs(const float* __restrict__ x, const float* __restrict__ Wrw,
        const float* __restrict__ brw, float* __restrict__ probs)
{
  const int tid = threadIdx.x;
  const int row = blockIdx.x * 32 + (tid >> 3);
  const int r   = tid & 7;
  const float* xr = x + (long)row * D_;
  const float* wr = Wrw + (long)r * D_;
  float dot = brw[r];
  for (int d = 0; d < D_; ++d) dot += xr[d] * wr[d];
  float m = dot;
  for (int o = 1; o < 8; o <<= 1) m = fmaxf(m, __shfl_xor(m, o, 8));
  float e = expf(dot - m), s = e;
  for (int o = 1; o < 8; o <<= 1) s += __shfl_xor(s, o, 8);
  probs[(long)row * R_ + r] = e / s;
}

__global__ void k_axpy(float* __restrict__ x, const float* __restrict__ y,
                       float a, long n)
{
  long i = (long)blockIdx.x * blockDim.x + threadIdx.x;
  if (i < n) x[i] += a * y[i];
}

__global__ void __launch_bounds__(256)
k_tension(const float* __restrict__ f, float* __restrict__ t)
{
  __shared__ float red[256];
  const int row = blockIdx.x, tid = threadIdx.x;
  const float* x = f + (long)row * D_;
  float s = 0.f;
  for (int d = tid; d < D_; d += 256) { float v = x[d]; s += v * v; }
  red[tid] = s; __syncthreads();
  for (int o = 128; o > 0; o >>= 1) { if (tid < o) red[tid] += red[tid + o]; __syncthreads(); }
  if (tid == 0) t[row] = red[0] / (float)D_;
}

// x += fout (+ GATE * tension_prev[row] * W_tension[d])
__global__ void k_xupdate(float* __restrict__ x, const float* __restrict__ f,
                          const float* __restrict__ tprev,
                          const float* __restrict__ Wt, long n)
{
  long i = (long)blockIdx.x * blockDim.x + threadIdx.x;
  if (i >= n) return;
  long row = i / D_; int d = (int)(i % D_);
  float add = f[i];
  if (tprev) add += GATE_ * tprev[row] * Wt[d];
  x[i] += add;
}

// ---------------- host-side GEMM dispatcher ----------------
static void gemm(hipStream_t st,
                 const float* A, long lda, long sA,
                 const float* W, long ldw, long sW,
                 const float* bias, float* C, long ldc, long sC,
                 int M, int N, int K, int batch, int accum, float alpha,
                 const float* rs, int rsStride, int act, int transb)
{
  dim3 g((unsigned)((N + 127) / 128), (unsigned)(M / 64), (unsigned)batch), b(256);
  if      (act == 0 && transb == 0) k_gemm<0,0><<<g,b,0,st>>>(A,lda,sA,W,ldw,sW,bias,C,ldc,sC,M,N,K,accum,alpha,rs,rsStride);
  else if (act == 1 && transb == 0) k_gemm<1,0><<<g,b,0,st>>>(A,lda,sA,W,ldw,sW,bias,C,ldc,sC,M,N,K,accum,alpha,rs,rsStride);
  else if (act == 0 && transb == 1) k_gemm<0,1><<<g,b,0,st>>>(A,lda,sA,W,ldw,sW,bias,C,ldc,sC,M,N,K,accum,alpha,rs,rsStride);
  else                              k_gemm<1,1><<<g,b,0,st>>>(A,lda,sA,W,ldw,sW,bias,C,ldc,sC,M,N,K,accum,alpha,rs,rsStride);
}

extern "C" void kernel_launch(void* const* d_in, const int* in_sizes, int n_in,
                              void* d_out, int out_size, void* d_ws, size_t ws_size,
                              hipStream_t stream)
{
  (void)in_sizes; (void)n_in; (void)out_size; (void)ws_size;

  const int*   idx   = (const int*)  d_in[0];
  const float* tok   = (const float*)d_in[1];
  const float* pos   = (const float*)d_in[2];
  const float* ln1w  = (const float*)d_in[3];
  const float* ln1b  = (const float*)d_in[4];
  const float* Wattn = (const float*)d_in[5];
  const float* battn = (const float*)d_in[6];
  const float* Wproj = (const float*)d_in[7];
  const float* bproj = (const float*)d_in[8];
  const float* Wca   = (const float*)d_in[9];
  const float* Wrw   = (const float*)d_in[10];
  const float* brw   = (const float*)d_in[11];
  const float* Wrul  = (const float*)d_in[12];
  const float* lncaw = (const float*)d_in[13];
  const float* lncab = (const float*)d_in[14];
  const float* ln2w  = (const float*)d_in[15];
  const float* ln2b  = (const float*)d_in[16];
  const float* Wa1   = (const float*)d_in[17];
  const float* ba1   = (const float*)d_in[18];
  const float* Wa2   = (const float*)d_in[19];
  const float* ba2   = (const float*)d_in[20];
  const float* Wg1   = (const float*)d_in[21];
  const float* bg1   = (const float*)d_in[22];
  const float* Wg2   = (const float*)d_in[23];
  const float* bg2   = (const float*)d_in[24];
  const float* Wt    = (const float*)d_in[25];
  const float* lnfw  = (const float*)d_in[26];
  const float* lnfb  = (const float*)d_in[27];
  const float* Whg   = (const float*)d_in[28];

  // ---- workspace carve-up (~240 MB) ----
  float* ws = (float*)d_ws;
  long o = 0;
  float* x      = ws + o; o += (long)M_ * D_;
  float* h      = ws + o; o += (long)M_ * D_;
  float* qkv    = ws + o; o += (long)M_ * D3_;
  float* scores = ws + o; o += (long)H_ * T_ * T_;   // one batch element at a time
  float* y      = ws + o; o += (long)M_ * D_;
  float* nb     = ws + o; o += (long)M_ * D3_;
  float* ca     = ws + o; o += (long)M_ * D_;
  float* meta   = ws + o; o += (long)M_ * D_;
  float* probs  = ws + o; o += (long)M_ * R_;
  float* mid    = ws + o; o += (long)M_ * D4_;
  float* fout   = ws + o; o += (long)M_ * D_;

  float* logits_a = (float*)d_out;
  float* logits_g = logits_a + (long)M_ * V_;
  float* tens     = logits_g + (long)M_ * V_;        // (L, B*T)

  const long nXD = (long)M_ * D_;
  const int  EB  = 256;
  const int  gXD = (int)((nXD + EB - 1) / EB);

  k_embed<<<gXD, EB, 0, stream>>>(idx, tok, pos, x);

  for (int l = 0; l < L_; ++l) {
    // ---- attention ----
    k_layernorm<<<M_, 256, 0, stream>>>(x, h, ln1w + (long)l * D_, ln1b + (long)l * D_, D_);
    gemm(stream, h, D_, 0, Wattn + (long)l * D3_ * D_, D_, 0, battn + (long)l * D3_,
         qkv, D3_, 0, M_, D3_, D_, 1, 0, 1.f, nullptr, 0, 0, 0);

    for (int bb = 0; bb < B_; ++bb) {
      const float* qb = qkv + (long)bb * T_ * D3_;
      // scores[h] = q_h @ k_h^T   (batch over heads via stride 64)
      gemm(stream, qb, D3_, HD_, qb + D_, D3_, HD_, nullptr,
           scores, T_, (long)T_ * T_, T_, T_, HD_, H_, 0, 1.f, nullptr, 0, 0, 0);
      k_softmax_causal<<<dim3(T_, H_), 256, 0, stream>>>(scores);
      // y_h = att_h @ v_h         (A @ W form, v rows along K with ldw=3D)
      gemm(stream, scores, T_, (long)T_ * T_, qb + 2 * D_, D3_, HD_, nullptr,
           y + (long)bb * T_ * D_, D_, HD_, T_, HD_, T_, H_, 0, 1.f, nullptr, 0, 0, 1);
    }
    // x += y @ W_proj^T + b_proj
    gemm(stream, y, D_, 0, Wproj + (long)l * D_ * D_, D_, 0, bproj + (long)l * D_,
         x, D_, 0, M_, D_, D_, 1, 1, 1.f, nullptr, 0, 0, 0);

    // ---- CA neighbor mixing + META-CA ----
    k_build_nb<<<(int)(((long)M_ * D3_ + EB - 1) / EB), EB, 0, stream>>>(x, nb);
    gemm(stream, nb, D3_, 0, Wca + (long)l * D_ * D3_, D3_, 0, nullptr,
         ca, D_, 0, M_, D_, D3_, 1, 0, 1.f, nullptr, 0, 0, 0);
    k_probs<<<M_ / 32, 256, 0, stream>>>(x, Wrw + (long)l * R_ * D_, brw + (long)l * R_, probs);
    for (int r = 0; r < R_; ++r)   // meta += probs_r * (ca @ W_rules[r]^T)
      gemm(stream, ca, D_, 0, Wrul + ((long)l * R_ + r) * D_ * D_, D_, 0, nullptr,
           meta, D_, 0, M_, D_, D_, 1, (r > 0), 1.f, probs + r, R_, 0, 0);
    k_axpy<<<gXD, EB, 0, stream>>>(x, meta, GATE_, nXD);
    k_layernorm<<<M_, 256, 0, stream>>>(x, x, lncaw + (long)l * D_, lncab + (long)l * D_, D_);

    // ---- PureField dual-engine FFN ----
    k_layernorm<<<M_, 256, 0, stream>>>(x, h, ln2w + (long)l * D_, ln2b + (long)l * D_, D_);
    gemm(stream, h, D_, 0, Wa1 + (long)l * D4_ * D_, D_, 0, ba1 + (long)l * D4_,
         mid, D4_, 0, M_, D4_, D_, 1, 0, 1.f, nullptr, 0, 1, 0);           // gelu
    gemm(stream, mid, D4_, 0, Wa2 + (long)l * D_ * D4_, D4_, 0, ba2 + (long)l * D_,
         fout, D_, 0, M_, D_, D4_, 1, 0, 1.f, nullptr, 0, 0, 0);           // a
    gemm(stream, h, D_, 0, Wg1 + (long)l * D4_ * D_, D_, 0, bg1 + (long)l * D4_,
         mid, D4_, 0, M_, D4_, D_, 1, 0, 1.f, nullptr, 0, 1, 0);           // gelu
    gemm(stream, mid, D4_, 0, Wg2 + (long)l * D_ * D4_, D4_, 0, bg2 + (long)l * D_,
         fout, D_, 0, M_, D_, D4_, 1, 1, -1.f, nullptr, 0, 0, 0);          // a - g

    k_tension<<<M_, 256, 0, stream>>>(fout, tens + (long)l * M_);
    k_xupdate<<<gXD, EB, 0, stream>>>(x, fout,
                                      l > 0 ? tens + (long)(l - 1) * M_ : nullptr,
                                      Wt, nXD);
  }

  // ---- final LN + two heads ----
  k_layernorm<<<M_, 256, 0, stream>>>(x, h, lnfw, lnfb, D_);
  gemm(stream, h, D_, 0, tok, D_, 0, nullptr, logits_a, V_, 0, M_, V_, D_, 1, 0, 1.f, nullptr, 0, 0, 0);
  gemm(stream, h, D_, 0, Whg, D_, 0, nullptr, logits_g, V_, 0, M_, V_, D_, 1, 0, 1.f, nullptr, 0, 0, 0);
}
